// MaskConv_2164663517536
// MI455X (gfx1250) — compile-verified, matrix operand order checked by reading, not run on hardware
//
#include <hip/hip_runtime.h>
#include <hip/hip_bf16.h>

// ---------------- problem constants (from reference) ----------------
#define N_NODES 50000
#define N_EDGES 1600000
#define C_INF   128
#define C_OUTF  128
#define ZR      0.8f

// 50000 = 16*3125 -> row tiles = 3125, col tiles = 8, total 25000 tiles, /4 waves = 6250 blocks
#define ROW_TILES 3125
#define GEMM_BLOCKS 6250

typedef __attribute__((ext_vector_type(16))) __bf16 v16bf;
typedef __attribute__((ext_vector_type(8)))  float  v8f;

union FragU { unsigned int u[8]; v16bf v; };

__device__ __forceinline__ unsigned short f32bf(float f) {
    unsigned int u = __float_as_uint(f);
    u += 0x7FFFu + ((u >> 16) & 1u);          // round-to-nearest-even
    return (unsigned short)(u >> 16);
}

// ---------------- small elementwise kernels ----------------
__global__ void k_cvt_bf16(const float* __restrict__ src, unsigned short* __restrict__ dst, int n) {
    int i = blockIdx.x * blockDim.x + threadIdx.x;
    if (i < n) dst[i] = f32bf(src[i]);
}

// cat[:,128:256] = bf16(x)
__global__ void k_fill_x(const float* __restrict__ x, unsigned short* __restrict__ cat) {
    int i = blockIdx.x * blockDim.x + threadIdx.x;
    if (i >= N_NODES * C_INF) return;
    int n = i >> 7, f = i & 127;
    cat[(size_t)n * 256 + 128 + f] = f32bf(x[i]);
}

__global__ void k_deg(const int* __restrict__ row, const float* __restrict__ ew,
                      float* __restrict__ deg) {
    int e = blockIdx.x * blockDim.x + threadIdx.x;
    if (e < N_EDGES) atomicAdd(&deg[row[e]], ew[e]);
}

__global__ void k_fix_deg(float* __restrict__ deg) {
    int i = blockIdx.x * blockDim.x + threadIdx.x;
    if (i < N_NODES) {
        float d = deg[i];
        deg[i] = (d < 0.5f) ? d + 1.0f : d;
    }
}

// ---------------- dual-branch WMMA GEMM ----------------
// out[n, col] = mix(mask[n], relu?(A@B0^T + b0), relu?(A@B1^T + b1))
// A: bf16 [N, LDA] row-major, tile columns start at AOFF. B*: bf16 [128, KDIM] row-major.
// One wave computes one 16x16 output tile; K consumed 32 at a time via v_wmma_f32_16x16x32_bf16.
template<int KDIM, bool RELU, int AOFF, int LDA>
__global__ __launch_bounds__(128) void k_gemm_dual(
    const unsigned short* __restrict__ A,
    const unsigned short* __restrict__ B0,
    const unsigned short* __restrict__ B1,
    const float* __restrict__ bias0, const float* __restrict__ bias1,
    const int* __restrict__ mask,
    float* __restrict__ out)
{
    const int lane   = threadIdx.x & 31;
    const int wave   = threadIdx.x >> 5;
    const int tile   = blockIdx.x * 4 + wave;          // 25000 tiles total
    const int rowT   = tile >> 3;                      // 8 col tiles (128/16)
    const int colT   = tile & 7;
    const int laneHi = lane >> 4;
    const int lane16 = lane & 15;

    const unsigned short* arow  = A  + (size_t)(rowT * 16 + lane16) * LDA + AOFF;
    const unsigned short* b0row = B0 + (size_t)(colT * 16 + lane16) * KDIM;
    const unsigned short* b1row = B1 + (size_t)(colT * 16 + lane16) * KDIM;
    const int khalf = laneHi ? 8 : 0;

    v8f acc0 = {}; v8f acc1 = {};
#pragma unroll
    for (int kk = 0; kk < KDIM; kk += 32) {
        FragU fa, fb0, fb1;
#pragma unroll
        for (int v = 0; v < 8; ++v) {
            // 16-bit A layout: VGPR v holds K pair (k,k+1); +16 for VGPRs 4..7; +8 for upper half-wave
            int k = kk + 2 * (v & 3) + ((v & 4) ? 16 : 0) + khalf;
            fa.u[v]  = *(const unsigned int*)(arow  + k);
            fb0.u[v] = *(const unsigned int*)(b0row + k);
            fb1.u[v] = *(const unsigned int*)(b1row + k);
        }
        acc0 = __builtin_amdgcn_wmma_f32_16x16x32_bf16(false, fa.v, false, fb0.v,
                                                       (short)0, acc0, false, false);
        acc1 = __builtin_amdgcn_wmma_f32_16x16x32_bf16(false, fa.v, false, fb1.v,
                                                       (short)0, acc1, false, false);
    }

    // epilogue: C/D layout -> element (M = r + 8*laneHi, N = lane16)
    const int col = colT * 16 + lane16;
    const float bb0 = bias0[col], bb1 = bias1[col];
#pragma unroll
    for (int r = 0; r < 8; ++r) {
        const int nrow = rowT * 16 + r + 8 * laneHi;
        float o0 = acc0[r] + bb0;
        float o1 = acc1[r] + bb1;
        if (RELU) { o0 = fmaxf(o0, 0.0f); o1 = fmaxf(o1, 0.0f); }
        const bool msk = mask[nrow] != 0;
        const float res = msk ? (ZR * o1 + (1.0f - ZR) * o0)
                              : (ZR * o0 + (1.0f - ZR) * o1);
        out[(size_t)nrow * C_OUTF + col] = res;
    }
}

// ---------------- edge scatter: agg[row] += (ew/deg[row]) * h[col] ----------------
// One wave per edge; each lane handles a float4 of the 128 features. L2-resident.
__global__ __launch_bounds__(256) void k_edge(const int* __restrict__ row,
                                              const int* __restrict__ col,
                                              const float* __restrict__ ew,
                                              const float* __restrict__ deg,
                                              const float* __restrict__ h,
                                              float* __restrict__ agg)
{
    const int e = blockIdx.x * 8 + (threadIdx.x >> 5);
    const int lane = threadIdx.x & 31;
    if (e >= N_EDGES) return;
    const int r = row[e], c = col[e];
    const float w = ew[e] / deg[r];
    const float4 hv = ((const float4*)(h + (size_t)c * 128))[lane];
    float* dst = agg + (size_t)r * 128 + lane * 4;
    atomicAdd(dst + 0, w * hv.x);
    atomicAdd(dst + 1, w * hv.y);
    atomicAdd(dst + 2, w * hv.z);
    atomicAdd(dst + 3, w * hv.w);
}

// ---------------- GraphNorm stats (one pass: sum + sumsq per feature) ----------------
__global__ __launch_bounds__(128) void k_stats(const float* __restrict__ agg,
                                               float* __restrict__ stats)
{
    const int f = threadIdx.x;                 // 128 features
    const int base = blockIdx.x * 200;         // 250 blocks * 200 rows = 50000
    float s = 0.0f, s2 = 0.0f;
    for (int i = 0; i < 200; ++i) {
        float v = agg[(size_t)(base + i) * 128 + f];
        s += v; s2 += v * v;
    }
    atomicAdd(&stats[f], s);
    atomicAdd(&stats[128 + f], s2);
}

__global__ void k_finalize(const float* __restrict__ stats,
                           const float* __restrict__ gms,
                           const float* __restrict__ gnw,
                           float* __restrict__ muf, float* __restrict__ scl)
{
    const int f = threadIdx.x;                 // 128 threads, 1 block
    const float inv_n = 1.0f / (float)N_NODES;
    const float mean = stats[f] * inv_n;
    const float mu   = mean * gms[f];
    const float ex2  = stats[128 + f] * inv_n;
    // E[(agg-mu)^2] = E[agg^2] - 2*mu*E[agg] + mu^2
    const float var  = ex2 - 2.0f * mu * mean + mu * mu;
    muf[f] = mu;
    scl[f] = gnw[f] * rsqrtf(var + 1e-5f);
}

// g = scl*(agg - mu) + gn_bias -> bf16 into cat[:,0:128]
__global__ void k_norm(const float* __restrict__ agg, const float* __restrict__ muf,
                       const float* __restrict__ scl, const float* __restrict__ gnb,
                       unsigned short* __restrict__ cat)
{
    int i = blockIdx.x * blockDim.x + threadIdx.x;
    if (i >= N_NODES * 128) return;
    int n = i >> 7, f = i & 127;
    float g = scl[f] * (agg[i] - muf[f]) + gnb[f];
    cat[(size_t)n * 256 + f] = f32bf(g);
}

// ---------------- launch ----------------
extern "C" void kernel_launch(void* const* d_in, const int* in_sizes, int n_in,
                              void* d_out, int out_size, void* d_ws, size_t ws_size,
                              hipStream_t stream) {
    (void)in_sizes; (void)n_in; (void)out_size; (void)ws_size;
    const float* x    = (const float*)d_in[0];
    const int*   eidx = (const int*)  d_in[1];   // [2, E] flat
    const float* ew   = (const float*)d_in[2];
    const int*   mask = (const int*)  d_in[3];
    const float* Wt0  = (const float*)d_in[4];
    const float* bt0  = (const float*)d_in[5];
    const float* Wt1  = (const float*)d_in[6];
    const float* bt1  = (const float*)d_in[7];
    const float* Wc0  = (const float*)d_in[8];
    const float* bc0  = (const float*)d_in[9];
    const float* Wc1  = (const float*)d_in[10];
    const float* bc1  = (const float*)d_in[11];
    const float* gnw  = (const float*)d_in[12];
    const float* gnb  = (const float*)d_in[13];
    const float* gms  = (const float*)d_in[14];
    float* out = (float*)d_out;

    const int* erow = eidx;
    const int* ecol = eidx + N_EDGES;

    // carve workspace (256B aligned)
    char* base = (char*)d_ws;
    size_t off = 0;
    auto carve = [&](size_t bytes) { char* p = base + off; off = (off + bytes + 255) & ~(size_t)255; return p; };
    float*          deg   = (float*)         carve((size_t)N_NODES * 4);
    float*          agg   = (float*)         carve((size_t)N_NODES * 128 * 4);
    float*          h     = (float*)         carve((size_t)N_NODES * 128 * 4);
    unsigned short* cat   = (unsigned short*)carve((size_t)N_NODES * 256 * 2);
    unsigned short* wt0b  = (unsigned short*)carve((size_t)128 * 128 * 2);
    unsigned short* wt1b  = (unsigned short*)carve((size_t)128 * 128 * 2);
    unsigned short* wc0b  = (unsigned short*)carve((size_t)128 * 256 * 2);
    unsigned short* wc1b  = (unsigned short*)carve((size_t)128 * 256 * 2);
    float*          stats = (float*)         carve(256 * 4);
    float*          muf   = (float*)         carve(128 * 4);
    float*          scl   = (float*)         carve(128 * 4);

    // zero accumulators every call (graph replay does not re-poison)
    hipMemsetAsync(deg,   0, (size_t)N_NODES * 4,        stream);
    hipMemsetAsync(agg,   0, (size_t)N_NODES * 128 * 4,  stream);
    hipMemsetAsync(stats, 0, 256 * 4,                    stream);

    // fp32 -> bf16 weight/activation conversion
    k_cvt_bf16<<<64,  256, 0, stream>>>(Wt0, wt0b, 128 * 128);
    k_cvt_bf16<<<64,  256, 0, stream>>>(Wt1, wt1b, 128 * 128);
    k_cvt_bf16<<<128, 256, 0, stream>>>(Wc0, wc0b, 128 * 256);
    k_cvt_bf16<<<128, 256, 0, stream>>>(Wc1, wc1b, 128 * 256);
    k_fill_x<<<(N_NODES * 128 + 255) / 256, 256, 0, stream>>>(x, cat);

    // degree normalization
    k_deg<<<N_EDGES / 256, 256, 0, stream>>>(erow, ew, deg);
    k_fix_deg<<<(N_NODES + 255) / 256, 256, 0, stream>>>(deg);

    // GEMM 1: h = mix(mask, relu(x@Wt0^T+bt0), relu(x@Wt1^T+bt1))
    k_gemm_dual<128, true, 128, 256><<<GEMM_BLOCKS, 128, 0, stream>>>(
        cat, wt0b, wt1b, bt0, bt1, mask, h);

    // sparse aggregate
    k_edge<<<N_EDGES / 8, 256, 0, stream>>>(erow, ecol, ew, deg, h, agg);

    // GraphNorm
    k_stats<<<250, 128, 0, stream>>>(agg, stats);
    k_finalize<<<1, 128, 0, stream>>>(stats, gms, gnw, muf, scl);
    k_norm<<<(N_NODES * 128 + 255) / 256, 256, 0, stream>>>(agg, muf, scl, gnb, cat);

    // GEMM 2: out = mix(mask, cat@Wc0^T+bc0, cat@Wc1^T+bc1)
    k_gemm_dual<256, false, 0, 256><<<GEMM_BLOCKS, 128, 0, stream>>>(
        cat, wc0b, wc1b, bc0, bc1, mask, out);
}
